// mBP_model_88613765251381
// MI455X (gfx1250) — matrix operand order repaired and checked by reading, not compiled, not run on hardware
//
#include <hip/hip_runtime.h>
#include <hip/hip_bf16.h>

#define NAT  4096
#define KNB  64
#define EDG  (NAT*KNB)
#define NRAD 32
#define RCUT 5.0f
#define PIF  3.14159265358979323846f
#define NITER 32
#define RBLK 64

typedef _Float16 half_t;
typedef __attribute__((ext_vector_type(16))) _Float16 v16h;
typedef __attribute__((ext_vector_type(8)))  _Float16 v8h;
typedef __attribute__((ext_vector_type(8)))  float    v8f;

__device__ __forceinline__ v8f wmma16(v16h a, v16h b, v8f c) {
  return __builtin_amdgcn_wmma_f32_16x16x32_f16(false, a, false, b, (short)0, c, false, false);
}

__device__ __forceinline__ float silu_f(float x) { return x / (1.0f + __expf(-x)); }

__device__ __forceinline__ v16h cat8(v8h lo, v8h hh) {
  return __builtin_shufflevector(lo, hh, 0, 1, 2, 3, 4, 5, 6, 7,
                                         8, 9, 10, 11, 12, 13, 14, 15);
}

// A fragment 16x32 f16 from row-major [16 x ld] (ld multiple of 8, 16B-aligned base).
// ISA layout: lane = hi*16 + m; elements 0..7 -> K = hi*8 + 0..7, 8..15 -> K = hi*8+16..23
// => two contiguous 8-half runs per lane: b128 loads.
__device__ __forceinline__ v16h loadA(const half_t* base, int ld, int lane) {
  int m = lane & 15, hi = lane >> 4;
  const half_t* rp = base + m * ld + hi * 8;
  v8h lo = *(const v8h*)rp;
  v8h hh = *(const v8h*)(rp + 16);
  return cat8(lo, hh);
}

// B fragment from pre-swizzled fragment-major weights: P[tile][lane][16 halfs]
__device__ __forceinline__ v16h loadBp(const half_t* P, int tile, int lane) {
  const half_t* rp = P + (((size_t)tile * 32) + lane) * 16;
  v8h lo = *(const v8h*)rp;
  v8h hh = *(const v8h*)(rp + 8);
  return cat8(lo, hh);
}

// ---------------- weight conversion f32 -> f16 fragment-major tiles ----------------
// B-tile layout (32x16): lane = kh*16 + n, element j holds W[kc*32 + (j<8 ? kh*8+j : 16+kh*8+j-8)][tn*16 + n]
__device__ __forceinline__ void pack_w(const float* W, half_t* P, int K, int N, int t, int st) {
  int total = K * N;
  int ntn = N >> 4;
  for (int idx = t; idx < total; idx += st) {
    int k = idx / N, n = idx - k * N;
    int kc = k >> 5, kk = k & 31;
    int kh = (kk >> 3) & 1, j = (kk & 7) + ((kk >> 4) << 3);
    int tn = n >> 4, nn = n & 15;
    int tile = kc * ntn + tn;
    P[(((size_t)tile * 32) + kh * 16 + nn) * 16 + j] = (half_t)W[idx];
  }
}

__global__ void k_cvt(const float* RW1, const float* RW2, const float* AW1, const float* AW2,
                      half_t* RW1p, half_t* RW2p, half_t* AW1p, half_t* AW2p) {
  int t = blockIdx.x * blockDim.x + threadIdx.x;
  int st = gridDim.x * blockDim.x;
  pack_w(RW1, RW1p, 32, 64, t, st);
  pack_w(RW2, RW2p, 64, 128, t, st);
  pack_w(AW1, AW1p, 512, 128, t, st);
  pack_w(AW2, AW2p, 128, 128, t, st);
}

// ---------------- species encoder (tiny MLP, one-hot => row gather) ----------------
__global__ void k_species(const int* sid, const float* SW1, const float* Sb1,
                          const float* SW2, const float* Sb2, float* enc) {
  int i = blockIdx.x * blockDim.x + threadIdx.x;
  if (i >= NAT) return;
  int s = sid[i];
  float h[16];
#pragma unroll
  for (int j = 0; j < 16; ++j) h[j] = silu_f(SW1[s * 16 + j] + Sb1[j]);
#pragma unroll
  for (int c = 0; c < 4; ++c) {
    float a = Sb2[c];
#pragma unroll
    for (int j = 0; j < 16; ++j) a += h[j] * SW2[j * 4 + c];
    enc[i * 4 + c] = a;
  }
}

// ---------------- fused bessel basis + radial MLP (WMMA) ----------------
// one wave per 16-edge tile; bessel computed directly into A-fragment registers.
// Output stored per-atom transposed: rb0T[atom][nrad][64 edges] for the desc GEMM.
__global__ void __launch_bounds__(128) k_radial(
    const float* pos, const float* cell, const int* fidx, const int* sidx, const int* shv,
    const half_t* RW1p, const float* Rb1, const half_t* RW2p, const float* Rb2,
    half_t* rb0T) {
  __shared__ __align__(16) half_t hbuf[4][16 * 64];
  int lane = threadIdx.x & 31;
  int wave = threadIdx.x >> 5;
  int e0 = (blockIdx.x * 4 + wave) * 16;
  int m = lane & 15, hi = lane >> 4;
  int e = e0 + m;

  int ia = fidx[e], ja = sidx[e];
  float sx = (float)shv[e * 3 + 0], sy = (float)shv[e * 3 + 1], sz = (float)shv[e * 3 + 2];
  float dx = pos[ja * 3 + 0] - pos[ia * 3 + 0] + sx * cell[0] + sy * cell[3] + sz * cell[6];
  float dy = pos[ja * 3 + 1] - pos[ia * 3 + 1] + sx * cell[1] + sy * cell[4] + sz * cell[7];
  float dz = pos[ja * 3 + 2] - pos[ia * 3 + 2] + sx * cell[2] + sy * cell[5] + sz * cell[8];
  float r = sqrtf(dx * dx + dy * dy + dz * dz);
  float fc = (r < RCUT) ? 0.5f * (__cosf(PIF * r / RCUT) + 1.0f) : 0.0f;
  float pref = sqrtf(2.0f / RCUT) * fc / (r + 1e-12f);

  // A fragment of bessel basis (each lane computes exactly the K it owns)
  v16h a;
#pragma unroll
  for (int t = 0; t < 16; ++t) {
    int k = (t < 8 ? t : t + 8) + hi * 8;   // K index 0..31
    float nn = (float)(k + 1);
    a[t] = (half_t)(pref * __sinf(nn * PIF * r / RCUT));
  }

  // GEMM1: [16,32] x RW1[32,64] -> silu -> LDS (f16)
#pragma unroll
  for (int t = 0; t < 4; ++t) {
    v16h b = loadBp(RW1p, t, lane);
    v8f c = {0, 0, 0, 0, 0, 0, 0, 0};
    c = wmma16(a, b, c);
    int ncol = t * 16 + m;
    float bb = Rb1[ncol];
#pragma unroll
    for (int jj = 0; jj < 8; ++jj)
      hbuf[wave][(jj + hi * 8) * 64 + ncol] = (half_t)silu_f(c[jj] + bb);
  }
  __syncthreads();

  // GEMM2: [16,64] x RW2[64,128] -> silu -> component 0 -> rb0T
  v8f acc[8];
#pragma unroll
  for (int t = 0; t < 8; ++t) acc[t] = (v8f){0, 0, 0, 0, 0, 0, 0, 0};
#pragma unroll
  for (int kc = 0; kc < 2; ++kc) {
    v16h a2 = loadA(&hbuf[wave][kc * 32], 64, lane);
#pragma unroll
    for (int tn = 0; tn < 8; ++tn) {
      v16h b = loadBp(RW2p, kc * 8 + tn, lane);
      acc[tn] = wmma16(a2, b, acc[tn]);
    }
  }
  int atom = e0 >> 6;
  int eb = e0 & 63;                          // edge offset of this tile within the atom
#pragma unroll
  for (int tn = 0; tn < 8; ++tn) {
    int ncol = tn * 16 + m;
    if ((ncol & 3) == 0) {                   // reshape(E,NRAD,NCOMP)[:,:,0]
      int n = ncol >> 2;
      float bb = Rb2[ncol];
#pragma unroll
      for (int jj = 0; jj < 8; ++jj)
        rb0T[(size_t)atom * 2048 + n * 64 + (eb + jj + hi * 8)] =
            (half_t)silu_f(acc[tn][jj] + bb);
    }
  }
}

// ---------------- per-atom descriptor: desc_i = rb0_i^T @ spec_i  (WMMA) ----------------
__global__ void __launch_bounds__(128) k_desc(
    const half_t* rb0T, const float* enc, const int* fidx, const int* sidx, half_t* desc) {
  int lane = threadIdx.x & 31;
  int wave = threadIdx.x >> 5;
  int atom = blockIdx.x * 4 + wave;
  size_t e0 = (size_t)atom * 64;
  int m = lane & 15, hi = lane >> 4;

  // B fragments: spec[e][s] = enc_i[s>>2]*enc_j[s&3], built on the fly
  v16h bfr[2];
#pragma unroll
  for (int kc = 0; kc < 2; ++kc) {
#pragma unroll
    for (int jj = 0; jj < 16; ++jj) {
      size_t e = e0 + (size_t)kc * 32 + hi * 8 + (jj < 8 ? jj : jj + 8);
      int i2 = fidx[e], j2 = sidx[e];
      bfr[kc][jj] = (half_t)(enc[i2 * 4 + (m >> 2)] * enc[j2 * 4 + (m & 3)]);
    }
  }
#pragma unroll
  for (int mt = 0; mt < 2; ++mt) {
    v8f acc = {0, 0, 0, 0, 0, 0, 0, 0};
#pragma unroll
    for (int kc = 0; kc < 2; ++kc) {
      // A[m_rad][k_edge] straight out of rb0T: two b128 runs per lane
      v16h a = loadA(rb0T + (size_t)atom * 2048 + (mt * 16 + m) * 64 - (size_t)m * 64 + kc * 32,
                     64, lane);
      acc = wmma16(a, bfr[kc], acc);
    }
#pragma unroll
    for (int jj = 0; jj < 8; ++jj) {
      int nr = mt * 16 + jj + hi * 8;
      desc[(size_t)atom * 512 + nr * 16 + m] = (half_t)acc[jj];
    }
  }
}

// ---------------- atomic energy net: 512->128 silu ->128 silu ->1 (WMMA) ----------------
__global__ void __launch_bounds__(256) k_atomic(
    const half_t* desc, const half_t* AW1p, const float* Ab1,
    const half_t* AW2p, const float* Ab2, const float* AW3, const float* Ab3,
    float* eat) {
  __shared__ __align__(16) half_t a1buf[8][16 * 128];
  int lane = threadIdx.x & 31, wave = threadIdx.x >> 5;
  int a0 = (blockIdx.x * 8 + wave) * 16;
  int m = lane & 15, hi = lane >> 4;

  v8f acc[8];
#pragma unroll
  for (int t = 0; t < 8; ++t) acc[t] = (v8f){0, 0, 0, 0, 0, 0, 0, 0};
  for (int kc = 0; kc < 16; ++kc) {
    v16h a = loadA(desc + (size_t)a0 * 512 + kc * 32, 512, lane);
#pragma unroll
    for (int tn = 0; tn < 8; ++tn) {
      v16h b = loadBp(AW1p, kc * 8 + tn, lane);
      acc[tn] = wmma16(a, b, acc[tn]);
    }
  }
#pragma unroll
  for (int tn = 0; tn < 8; ++tn) {
    int ncol = tn * 16 + m;
    float bb = Ab1[ncol];
#pragma unroll
    for (int jj = 0; jj < 8; ++jj)
      a1buf[wave][(jj + hi * 8) * 128 + ncol] = (half_t)silu_f(acc[tn][jj] + bb);
  }
  __syncthreads();

  v8f acc2[8];
#pragma unroll
  for (int t = 0; t < 8; ++t) acc2[t] = (v8f){0, 0, 0, 0, 0, 0, 0, 0};
#pragma unroll
  for (int kc = 0; kc < 4; ++kc) {
    v16h a = loadA(&a1buf[wave][kc * 32], 128, lane);
#pragma unroll
    for (int tn = 0; tn < 8; ++tn) {
      v16h b = loadBp(AW2p, kc * 8 + tn, lane);
      acc2[tn] = wmma16(a, b, acc2[tn]);
    }
  }
  // final 128 -> 1 fold + cross-lane reduction (butterfly over the 16-lane half)
  float part[8];
#pragma unroll
  for (int jj = 0; jj < 8; ++jj) part[jj] = 0.f;
#pragma unroll
  for (int tn = 0; tn < 8; ++tn) {
    int ncol = tn * 16 + m;
    float w3 = AW3[ncol], bb = Ab2[ncol];
#pragma unroll
    for (int jj = 0; jj < 8; ++jj)
      part[jj] += silu_f(acc2[tn][jj] + bb) * w3;
  }
#pragma unroll
  for (int off = 1; off < 16; off <<= 1)
#pragma unroll
    for (int jj = 0; jj < 8; ++jj)
      part[jj] += __shfl_xor(part[jj], off, 32);
  if (m == 0) {
    float b3 = Ab3[0];
#pragma unroll
    for (int jj = 0; jj < 8; ++jj)
      eat[a0 + jj + hi * 8] = part[jj] + b3;
  }
}

// ---------------- CG for charge equilibration: A x = E1, A y = 1 ----------------
__global__ void k_cg_init(const float* E1, float* x, float* r, float* p, float* part) {
  __shared__ float sm0[256], sm1[256];
  float s0 = 0.f, s1 = 0.f;
  for (int i = blockIdx.x * 256 + threadIdx.x; i < NAT; i += gridDim.x * 256) {
    float b0 = E1[i], b1 = 1.0f;
    x[i] = 0.f; x[NAT + i] = 0.f;
    r[i] = b0;  r[NAT + i] = b1;
    p[i] = b0;  p[NAT + i] = b1;
    s0 += b0 * b0; s1 += b1 * b1;
  }
  sm0[threadIdx.x] = s0; sm1[threadIdx.x] = s1;
  __syncthreads();
  for (int off = 128; off > 0; off >>= 1) {
    if (threadIdx.x < off) { sm0[threadIdx.x] += sm0[threadIdx.x + off]; sm1[threadIdx.x] += sm1[threadIdx.x + off]; }
    __syncthreads();
  }
  if (threadIdx.x == 0) { part[blockIdx.x] = sm0[0]; part[RBLK + blockIdx.x] = sm1[0]; }
}

__global__ void k_fin_rs0(const float* part, float* scal) {
  if (threadIdx.x == 0) {
    float a = 0.f, b = 0.f;
    for (int i = 0; i < RBLK; ++i) { a += part[i]; b += part[RBLK + i]; }
    scal[0] = a; scal[1] = b;
  }
}

// Ap = Vij @ p + E2 .* p  (both RHS, one wave per row, coalesced b128 loads)
__global__ void __launch_bounds__(256) k_matvec(const float* V, const float* E2,
                                                const float* p, float* Ap) {
  int lane = threadIdx.x & 31, wave = threadIdx.x >> 5;
  int row = blockIdx.x * 8 + wave;
  const float* Vr = V + (size_t)row * NAT;
  float s0 = 0.f, s1 = 0.f;
  for (int c0 = 0; c0 < NAT; c0 += 128) {
    __builtin_prefetch(Vr + c0 + 1024, 0, 1);
    float4 v  = *(const float4*)(Vr + c0 + lane * 4);
    float4 q0 = *(const float4*)(p + c0 + lane * 4);
    float4 q1 = *(const float4*)(p + NAT + c0 + lane * 4);
    s0 += v.x * q0.x + v.y * q0.y + v.z * q0.z + v.w * q0.w;
    s1 += v.x * q1.x + v.y * q1.y + v.z * q1.z + v.w * q1.w;
  }
#pragma unroll
  for (int off = 1; off < 32; off <<= 1) { s0 += __shfl_xor(s0, off, 32); s1 += __shfl_xor(s1, off, 32); }
  if (lane == 0) {
    float e2 = E2[row];
    Ap[row]       = s0 + e2 * p[row];
    Ap[NAT + row] = s1 + e2 * p[NAT + row];
  }
}

__global__ void k_dot_partial(const float* a, const float* b, float* part, int n, int slot) {
  __shared__ float sm[256];
  float s = 0.f;
  for (int i = blockIdx.x * 256 + threadIdx.x; i < n; i += gridDim.x * 256) s += a[i] * b[i];
  sm[threadIdx.x] = s;
  __syncthreads();
  for (int off = 128; off > 0; off >>= 1) {
    if (threadIdx.x < off) sm[threadIdx.x] += sm[threadIdx.x + off];
    __syncthreads();
  }
  if (threadIdx.x == 0) part[slot * RBLK + blockIdx.x] = sm[0];
}

__global__ void k_fin_alpha(const float* part, float* scal) {
  if (threadIdx.x == 0) {
    float p0 = 0.f, p1 = 0.f;
    for (int i = 0; i < RBLK; ++i) { p0 += part[i]; p1 += part[RBLK + i]; }
    scal[2] = scal[0] / p0;
    scal[3] = scal[1] / p1;
  }
}

__global__ void k_update_xr(float* x, float* r, const float* p, const float* Ap,
                            const float* scal, float* part) {
  __shared__ float sm0[256], sm1[256];
  float a0 = scal[2], a1 = scal[3];
  float s0 = 0.f, s1 = 0.f;
  for (int i = blockIdx.x * 256 + threadIdx.x; i < NAT; i += gridDim.x * 256) {
    x[i] += a0 * p[i];
    x[NAT + i] += a1 * p[NAT + i];
    float r0 = r[i] - a0 * Ap[i];             r[i] = r0;       s0 += r0 * r0;
    float r1 = r[NAT + i] - a1 * Ap[NAT + i]; r[NAT + i] = r1; s1 += r1 * r1;
  }
  sm0[threadIdx.x] = s0; sm1[threadIdx.x] = s1;
  __syncthreads();
  for (int off = 128; off > 0; off >>= 1) {
    if (threadIdx.x < off) { sm0[threadIdx.x] += sm0[threadIdx.x + off]; sm1[threadIdx.x] += sm1[threadIdx.x + off]; }
    __syncthreads();
  }
  if (threadIdx.x == 0) { part[blockIdx.x] = sm0[0]; part[RBLK + blockIdx.x] = sm1[0]; }
}

__global__ void k_fin_beta(const float* part, float* scal) {
  if (threadIdx.x == 0) {
    float r0 = 0.f, r1 = 0.f;
    for (int i = 0; i < RBLK; ++i) { r0 += part[i]; r1 += part[RBLK + i]; }
    scal[4] = r0 / scal[0];
    scal[5] = r1 / scal[1];
    scal[0] = r0; scal[1] = r1;
  }
}

__global__ void k_update_p(float* p, const float* r, const float* scal) {
  float b0 = scal[4], b1 = scal[5];
  for (int i = blockIdx.x * 256 + threadIdx.x; i < NAT; i += gridDim.x * 256) {
    p[i] = r[i] + b0 * p[i];
    p[NAT + i] = r[NAT + i] + b1 * p[NAT + i];
  }
}

// ---------------- final: lambda, e_coul = 1.5*sum(E1*q), add sum(e_atomic) ----------------
__global__ void k_final_partials(const float* x, const float* E1, const float* eat, float* part) {
  __shared__ float sm[5][256];
  float s[5] = {0.f, 0.f, 0.f, 0.f, 0.f};
  for (int i = blockIdx.x * 256 + threadIdx.x; i < NAT; i += gridDim.x * 256) {
    float x0 = x[i], x1 = x[NAT + i], e1 = E1[i];
    s[0] += x0; s[1] += x1; s[2] += e1 * x0; s[3] += e1 * x1; s[4] += eat[i];
  }
#pragma unroll
  for (int k = 0; k < 5; ++k) sm[k][threadIdx.x] = s[k];
  __syncthreads();
  for (int off = 128; off > 0; off >>= 1) {
    if (threadIdx.x < off)
      for (int k = 0; k < 5; ++k) sm[k][threadIdx.x] += sm[k][threadIdx.x + off];
    __syncthreads();
  }
  if (threadIdx.x == 0)
    for (int k = 0; k < 5; ++k) part[k * RBLK + blockIdx.x] = sm[k][0];
}

__global__ void k_write_out(const float* part, float* out) {
  if (threadIdx.x == 0 && blockIdx.x == 0) {
    float s[5] = {0.f, 0.f, 0.f, 0.f, 0.f};
    for (int i = 0; i < RBLK; ++i)
      for (int k = 0; k < 5; ++k) s[k] += part[k * RBLK + i];
    float lam = s[0] / s[1];                  // Q_TOT = 0
    out[0] = s[4] + 1.5f * (s[2] - lam * s[3]);
  }
}

extern "C" void kernel_launch(void* const* d_in, const int* in_sizes, int n_in,
                              void* d_out, int out_size, void* d_ws, size_t ws_size,
                              hipStream_t stream) {
  (void)in_sizes; (void)n_in; (void)out_size; (void)ws_size;
  const float* pos  = (const float*)d_in[0];
  const float* cell = (const float*)d_in[1];
  const int*   sid  = (const int*)d_in[2];
  const int*   fidx = (const int*)d_in[3];
  const int*   sidx = (const int*)d_in[4];
  const int*   shv  = (const int*)d_in[5];
  const float* SW1  = (const float*)d_in[6];
  const float* Sb1  = (const float*)d_in[7];
  const float* SW2  = (const float*)d_in[8];
  const float* Sb2  = (const float*)d_in[9];
  const float* RW1  = (const float*)d_in[10];
  const float* Rb1  = (const float*)d_in[11];
  const float* RW2  = (const float*)d_in[12];
  const float* Rb2  = (const float*)d_in[13];
  const float* AW1  = (const float*)d_in[14];
  const float* Ab1  = (const float*)d_in[15];
  const float* AW2  = (const float*)d_in[16];
  const float* Ab2  = (const float*)d_in[17];
  const float* AW3  = (const float*)d_in[18];
  const float* Ab3  = (const float*)d_in[19];
  const float* Vij  = (const float*)d_in[20];
  const float* E1   = (const float*)d_in[21];
  const float* E2   = (const float*)d_in[22];

  char* w = (char*)d_ws;
  half_t* RW1p = (half_t*)w; w += 2048 * 2;        // 4 tiles  * 512 halfs
  half_t* RW2p = (half_t*)w; w += 8192 * 2;        // 16 tiles
  half_t* AW1p = (half_t*)w; w += 65536 * 2;       // 128 tiles
  half_t* AW2p = (half_t*)w; w += 16384 * 2;       // 32 tiles
  float*  enc  = (float*)w;  w += (size_t)NAT * 4 * 4;
  float*  eat  = (float*)w;  w += (size_t)NAT * 4;
  float*  xv   = (float*)w;  w += (size_t)2 * NAT * 4;
  float*  rv   = (float*)w;  w += (size_t)2 * NAT * 4;
  float*  pv   = (float*)w;  w += (size_t)2 * NAT * 4;
  float*  apv  = (float*)w;  w += (size_t)2 * NAT * 4;
  float*  scal = (float*)w;  w += 64 * 4;
  float*  part = (float*)w;  w += (size_t)8 * RBLK * 4;
  half_t* rb0T = (half_t*)w; w += (size_t)EDG * 32 * 2;   // [NAT][32][64] transposed
  half_t* desc = (half_t*)w; w += (size_t)NAT * 512 * 2;

  // dense pipeline (WMMA)
  k_cvt<<<64, 256, 0, stream>>>(RW1, RW2, AW1, AW2, RW1p, RW2p, AW1p, AW2p);
  k_species<<<NAT / 256, 256, 0, stream>>>(sid, SW1, Sb1, SW2, Sb2, enc);
  k_radial<<<EDG / 64, 128, 0, stream>>>(pos, cell, fidx, sidx, shv,
                                         RW1p, Rb1, RW2p, Rb2, rb0T);
  k_desc<<<NAT / 4, 128, 0, stream>>>(rb0T, enc, fidx, sidx, desc);
  k_atomic<<<NAT / 128, 256, 0, stream>>>(desc, AW1p, Ab1, AW2p, Ab2, AW3, Ab3, eat);

  // charge equilibration: CG on A = Vij + diag(E2), two RHS (E1 and ones)
  k_cg_init<<<RBLK, 256, 0, stream>>>(E1, xv, rv, pv, part);
  k_fin_rs0<<<1, 1, 0, stream>>>(part, scal);
  for (int it = 0; it < NITER; ++it) {
    k_matvec<<<NAT / 8, 256, 0, stream>>>(Vij, E2, pv, apv);
    k_dot_partial<<<RBLK, 256, 0, stream>>>(pv, apv, part, NAT, 0);
    k_dot_partial<<<RBLK, 256, 0, stream>>>(pv + NAT, apv + NAT, part, NAT, 1);
    k_fin_alpha<<<1, 1, 0, stream>>>(part, scal);
    k_update_xr<<<RBLK, 256, 0, stream>>>(xv, rv, pv, apv, scal, part);
    k_fin_beta<<<1, 1, 0, stream>>>(part, scal);
    k_update_p<<<RBLK, 256, 0, stream>>>(pv, rv, scal);
  }
  k_final_partials<<<RBLK, 256, 0, stream>>>(xv, E1, eat, part);
  k_write_out<<<1, 1, 0, stream>>>(part, (float*)d_out);
}
